// MultiHeadAttention_63032940036635
// MI455X (gfx1250) — matrix operand order repaired and checked by reading, not compiled, not run on hardware
//
#include <hip/hip_runtime.h>
#include <hip/hip_bf16.h>

// ---------------------------------------------------------------------------
// MultiHeadAttention (channel attention): d=32 per head, softmax over 32x32.
// Pipeline:
//   1) Wqkv f32 -> f16
//   2) GEMM1 (WMMA f16): qkv[b] = Wqkv @ x[b] + bqkv   (q rows pre-scaled)
//   3) S = q k^T over N (WMMA), softmax -> attn (32x32 per b,h)
//   4) M[b] = Wout @ blockdiag(attn[b])  (fuses attn@v and Wout@o)
//   5) GEMM2 (WMMA f16): out[b] = M[b] @ v[b] + bout   (f32 output)
// CDNA5 paths: v_wmma_f32_16x16x32_f16, global_load_async_to_lds_b128
// (ASYNCcnt), s_wait_asynccnt, global_prefetch_b8, ds_add_f32 reduction.
// ---------------------------------------------------------------------------

typedef __attribute__((ext_vector_type(16))) _Float16 v16h;
typedef __attribute__((ext_vector_type(8)))  _Float16 v8h;
typedef __attribute__((ext_vector_type(8)))  float    v8f;
typedef __attribute__((ext_vector_type(4)))  float    v4f;

#define WMMA_F16(a, b, c) \
  __builtin_amdgcn_wmma_f32_16x16x32_f16(false, (a), false, (b), (short)0, (c), false, false)

// Async global->LDS 16B copy (GV mode). The LDS byte offset is the low 32
// bits of the generic address (ISA 10.2).
__device__ __forceinline__ void async_copy_b128(const void* gptr, void* lptr) {
  unsigned lds_addr = (unsigned)(unsigned long long)lptr;
  asm volatile("global_load_async_to_lds_b128 %0, %1, off"
               :
               : "v"(lds_addr), "v"(gptr)
               : "memory");
}
__device__ __forceinline__ void wait_asynccnt0() {
  asm volatile("s_wait_asynccnt 0x0" ::: "memory");
}

__device__ __forceinline__ v16h make16(v8h lo, v8h hi) {
  v16h r;
#pragma unroll
  for (int i = 0; i < 8; ++i) { r[i] = lo[i]; r[i + 8] = hi[i]; }
  return r;
}

// A fragment (16x32 f16, row-major, row stride = `stride` elements).
// Lane holds two contiguous 8-half runs at k = hi*8 and k = 16 + hi*8.
__device__ __forceinline__ v16h fragA(const _Float16* base, long stride) {
  const int lane = threadIdx.x & 31;
  const _Float16* p = base + (long)(lane & 15) * stride + ((lane >> 4) & 1) * 8;
  v8h lo = *(const v8h*)(p);
  v8h hi = *(const v8h*)(p + 16);
  return make16(lo, hi);
}

// B fragment (32x16 f16) from an n-major (transposed) tile: tileT[n][k].
// Lane holds two contiguous 8-half runs at k = hi*8 and k = 16 + hi*8.
__device__ __forceinline__ v16h fragB(const _Float16* baseT, long stride) {
  const int lane = threadIdx.x & 31;
  const _Float16* p = baseT + (long)(lane & 15) * stride + ((lane >> 4) & 1) * 8;
  v8h lo = *(const v8h*)(p);
  v8h hi = *(const v8h*)(p + 16);
  return make16(lo, hi);
}

constexpr float QK_SCALE = 0.17677669529663687f;  // 1/sqrt(32)

// ---------------------------------------------------------------------------
// Tiled WMMA GEMM: C[z] = A[z] (Mdim x Kdim, f16) @ B[z] (Kdim x Ndim) + bias.
// Block tile 64(M) x 128(N), 8 waves (2x4), each wave a 32x32 tile (2x2 WMMA).
// A tile staged via async DMA to LDS; B tile converted/transposed by VALU.
// ---------------------------------------------------------------------------
template <bool B_F32, bool OUT_F16, bool QSCALE>
__global__ __launch_bounds__(256) void gemm_wmma(
    const _Float16* __restrict__ A, long aStride,
    const void* __restrict__ Bv, long bStride,
    const float* __restrict__ bias,
    void* __restrict__ Cv,
    int Mdim, int Kdim, int Ndim) {
  __shared__ __align__(16) _Float16 As[64 * 32];    // 4 KB, k-major
  __shared__ __align__(16) _Float16 Bt[128 * 32];   // 8 KB, n-major (transposed)

  const int t  = threadIdx.x;
  const int z  = blockIdx.z;
  const int m0 = blockIdx.x * 64;
  const int n0 = blockIdx.y * 128;

  const _Float16* Ab = A + (long)z * aStride;
  const float*    Bf = (const float*)Bv + (long)z * bStride;      // used if B_F32
  const _Float16* Bh = (const _Float16*)Bv + (long)z * bStride;   // used if !B_F32

  const int wave = t >> 5, lane = t & 31;
  const int wm = wave & 1, wn = wave >> 1;

  v8f acc[2][2] = {};

  const int arow = t >> 2;            // 0..63
  const int aseg = (t & 3) * 8;       // 0,8,16,24
  const int kp   = (t >> 4) * 2;      // 0..30 (even k pair)
  const int nl   = (t & 15) * 8;      // 0..120

  for (int kt = 0; kt < Kdim; kt += 32) {
    __syncthreads();  // previous-iteration fragment reads complete
    // Stage A tile (64x32 f16): async DMA, one b128 per lane.
    async_copy_b128(&Ab[(long)(m0 + arow) * Kdim + kt + aseg],
                    &As[arow * 32 + aseg]);
    // Stage B tile transposed: thread covers k pair (kp,kp+1) x 8 n's.
    if constexpr (B_F32) {
      const float* r0 = Bf + (long)(kt + kp) * Ndim + n0 + nl;
      const float* r1 = r0 + Ndim;
      v4f a0 = *(const v4f*)r0, a1 = *(const v4f*)(r0 + 4);
      v4f c0 = *(const v4f*)r1, c1 = *(const v4f*)(r1 + 4);
#pragma unroll
      for (int i = 0; i < 8; ++i) {
        union { _Float16 h[2]; unsigned u; } pk;
        pk.h[0] = (_Float16)((i < 4) ? a0[i] : a1[i - 4]);
        pk.h[1] = (_Float16)((i < 4) ? c0[i] : c1[i - 4]);
        *(unsigned*)&Bt[(nl + i) * 32 + kp] = pk.u;
      }
      if (kt + 32 < Kdim)  // warm L2/WGP$ for next k-chunk
        __builtin_prefetch(r0 + 32L * Ndim, 0, 0);
    } else {
      const _Float16* r0 = Bh + (long)(kt + kp) * Ndim + n0 + nl;
      v8h h0 = *(const v8h*)r0;
      v8h h1 = *(const v8h*)(r0 + Ndim);
#pragma unroll
      for (int i = 0; i < 8; ++i) {
        union { _Float16 h[2]; unsigned u; } pk;
        pk.h[0] = h0[i];
        pk.h[1] = h1[i];
        *(unsigned*)&Bt[(nl + i) * 32 + kp] = pk.u;
      }
      if (kt + 32 < Kdim)
        __builtin_prefetch(r0 + 32L * Ndim, 0, 0);
    }
    wait_asynccnt0();   // A-tile DMA landed in LDS
    __syncthreads();

    v16h a0 = fragA(&As[(wm * 32 + 0)  * 32], 32);
    v16h a1 = fragA(&As[(wm * 32 + 16) * 32], 32);
    v16h b0 = fragB(&Bt[(wn * 32 + 0)  * 32], 32);
    v16h b1 = fragB(&Bt[(wn * 32 + 16) * 32], 32);
    acc[0][0] = WMMA_F16(a0, b0, acc[0][0]);
    acc[0][1] = WMMA_F16(a0, b1, acc[0][1]);
    acc[1][0] = WMMA_F16(a1, b0, acc[1][0]);
    acc[1][1] = WMMA_F16(a1, b1, acc[1][1]);
  }

  // Epilogue. C/D layout: N = lane&15; VGPR r -> M = r + 8*(lane>>4).
  const int cn = lane & 15;
  const int cm = ((lane >> 4) & 1) * 8;
#pragma unroll
  for (int mi = 0; mi < 2; ++mi)
#pragma unroll
    for (int ni = 0; ni < 2; ++ni)
#pragma unroll
      for (int r = 0; r < 8; ++r) {
        const int gm = m0 + wm * 32 + mi * 16 + cm + r;
        const int gn = n0 + wn * 32 + ni * 16 + cn;
        float v = acc[mi][ni][r] + bias[gm];
        if (QSCALE && gm < 256) v *= QK_SCALE;   // pre-scale q rows
        const long idx = (long)z * Mdim * (long)Ndim + (long)gm * Ndim + gn;
        if constexpr (OUT_F16) ((_Float16*)Cv)[idx] = (_Float16)v;
        else                   ((float*)Cv)[idx] = v;
      }
}

// ---------------------------------------------------------------------------
// S = q k^T over N (K=16384) + softmax. One block per (b,h); 8 waves split K.
// q rows are the WMMA A tile; k's row-major storage IS the n-major B layout.
// ---------------------------------------------------------------------------
__global__ __launch_bounds__(256) void attn_scores(
    const _Float16* __restrict__ qkv, float* __restrict__ attn, int Ndim) {
  __shared__ float S[1024];
  const int bh = blockIdx.x, b = bh >> 3, h = bh & 7;
  const int t = threadIdx.x, wave = t >> 5, lane = t & 31;
  const _Float16* qb = qkv + (long)b * 768 * Ndim + (long)(h * 32) * Ndim;
  const _Float16* kb = qkv + (long)b * 768 * Ndim + (long)(256 + h * 32) * Ndim;

  for (int i = t; i < 1024; i += 256) S[i] = 0.f;
  __syncthreads();

  v8f acc[2][2] = {};
  const int iters = Ndim / 256;  // 8 waves * 32 K per wave per iter
#pragma unroll 4
  for (int it = 0; it < iters; ++it) {
    const int nn0 = it * 256 + wave * 32;
    v16h a0 = fragA(qb + nn0, Ndim);                 // d = 0..15
    v16h a1 = fragA(qb + 16L * Ndim + nn0, Ndim);    // d = 16..31
    v16h b0 = fragB(kb + nn0, Ndim);                 // e = 0..15
    v16h b1 = fragB(kb + 16L * Ndim + nn0, Ndim);    // e = 16..31
    acc[0][0] = WMMA_F16(a0, b0, acc[0][0]);
    acc[0][1] = WMMA_F16(a0, b1, acc[0][1]);
    acc[1][0] = WMMA_F16(a1, b0, acc[1][0]);
    acc[1][1] = WMMA_F16(a1, b1, acc[1][1]);
  }

  // Cross-wave reduction into LDS (ds_add_f32).
  const int cn = lane & 15;
  const int cm = ((lane >> 4) & 1) * 8;
#pragma unroll
  for (int mi = 0; mi < 2; ++mi)
#pragma unroll
    for (int ni = 0; ni < 2; ++ni)
#pragma unroll
      for (int r = 0; r < 8; ++r)
        atomicAdd(&S[(mi * 16 + cm + r) * 32 + (ni * 16 + cn)], acc[mi][ni][r]);
  __syncthreads();

  // Row softmax: 32 lanes own one row each (q already scaled by 1/sqrt(d)).
  if (t < 32) {
    float mx = -3.4e38f;
    for (int e = 0; e < 32; ++e) mx = fmaxf(mx, S[t * 32 + e]);
    float sum = 0.f;
    for (int e = 0; e < 32; ++e) {
      float v = __expf(S[t * 32 + e] - mx);
      S[t * 32 + e] = v;
      sum += v;
    }
    const float inv = 1.f / sum;
    for (int e = 0; e < 32; ++e)
      attn[(long)bh * 1024 + t * 32 + e] = S[t * 32 + e] * inv;
  }
}

// ---------------------------------------------------------------------------
// M[b] = Wout @ blockdiag(attn[b]):  M[oo, h*32+e] = sum_d Wout[oo,h*32+d]*attn[b,h,d,e]
// ---------------------------------------------------------------------------
__global__ __launch_bounds__(256) void mix_kernel(
    const float* __restrict__ Wout, const float* __restrict__ attn,
    _Float16* __restrict__ M) {
  const int b = blockIdx.y, oo = blockIdx.x, c = threadIdx.x;  // c = 0..255
  const int h = c >> 5, e = c & 31;
  const float* arow = attn + ((long)(b * 8 + h) * 32) * 32 + e;  // stride 32 over d
  const float* wrow = Wout + (long)oo * 256 + h * 32;
  float s = 0.f;
#pragma unroll
  for (int d = 0; d < 32; ++d) s += wrow[d] * arow[d * 32];
  M[((long)b * 256 + oo) * 256 + c] = (_Float16)s;
}

__global__ __launch_bounds__(256) void cvt_f32_f16(
    const float* __restrict__ src, _Float16* __restrict__ dst, int n) {
  const int i = blockIdx.x * 256 + threadIdx.x;
  if (i < n) dst[i] = (_Float16)src[i];
}

// ---------------------------------------------------------------------------
extern "C" void kernel_launch(void* const* d_in, const int* in_sizes, int n_in,
                              void* d_out, int out_size, void* d_ws, size_t ws_size,
                              hipStream_t stream) {
  const float* x    = (const float*)d_in[0];
  const float* Wqkv = (const float*)d_in[1];
  const float* bqkv = (const float*)d_in[2];
  const float* Wout = (const float*)d_in[3];
  const float* bout = (const float*)d_in[4];
  float* out = (float*)d_out;
  (void)in_sizes; (void)n_in; (void)out_size; (void)ws_size;

  constexpr int  Bn = 8, C = 256, O3 = 768, HEADS = 8;
  constexpr int  N  = 128 * 128;  // 16384

  char* p = (char*)d_ws;
  auto alloc = [&](size_t bytes) { char* r = p; p += (bytes + 255) & ~(size_t)255; return r; };
  _Float16* Wq_h = (_Float16*)alloc((size_t)O3 * C * 2);           // 384 KB
  _Float16* qkv  = (_Float16*)alloc((size_t)Bn * O3 * N * 2);      // ~201 MB
  float*    attn = (float*)alloc((size_t)Bn * HEADS * 32 * 32 * 4);
  _Float16* Mx   = (_Float16*)alloc((size_t)Bn * C * C * 2);       // 1 MB

  // 1) weight convert
  cvt_f32_f16<<<(O3 * C + 255) / 256, 256, 0, stream>>>(Wqkv, Wq_h, O3 * C);

  // 2) qkv = Wqkv @ x + bqkv   (x is f32 B-matrix, converted in staging)
  gemm_wmma<true, true, true><<<dim3(O3 / 64, N / 128, Bn), 256, 0, stream>>>(
      Wq_h, 0L, (const void*)x, (long)C * N, bqkv, (void*)qkv, O3, C, N);

  // 3) attention scores + softmax
  attn_scores<<<Bn * HEADS, 256, 0, stream>>>(qkv, attn, N);

  // 4) M[b] = Wout @ blockdiag(attn[b])
  mix_kernel<<<dim3(C, Bn), 256, 0, stream>>>(Wout, attn, Mx);

  // 5) out = M @ v + bout  (v = qkv rows 512..767 per batch)
  gemm_wmma<false, false, false><<<dim3(C / 64, N / 128, Bn), 256, 0, stream>>>(
      Mx, (long)C * C, (const void*)(qkv + (size_t)512 * N), (long)O3 * N,
      bout, (void*)out, C, C, N);
}
